// CHiLAPModel_81509889344062
// MI455X (gfx1250) — matrix operand order repaired and checked by reading, not compiled, not run on hardware
//
#include <hip/hip_runtime.h>
#include <cstdint>
#include <cstddef>

typedef _Float16 half_t;
typedef __attribute__((ext_vector_type(16))) _Float16 v16h;
typedef __attribute__((ext_vector_type(8)))  _Float16 v8h;
typedef __attribute__((ext_vector_type(8)))  float    v8f;
typedef int v4i __attribute__((vector_size(16)));

#define DEVFN static __device__ __forceinline__

#if defined(__HIP_DEVICE_COMPILE__) && \
    __has_builtin(__builtin_amdgcn_global_load_async_to_lds_b128) && \
    __has_builtin(__builtin_amdgcn_s_wait_asynccnt)
#define USE_ASYNC_LDS 1
#else
#define USE_ASYNC_LDS 0
#endif

DEVFN v8f wmma_f16(v16h a, v16h b, v8f c) {
  return __builtin_amdgcn_wmma_f32_16x16x32_f16(false, a, false, b, (short)0, c, false, false);
}

DEVFN v16h join8(v8h lo, v8h hi) {
  return __builtin_shufflevector(lo, hi, 0, 1, 2, 3, 4, 5, 6, 7, 8, 9, 10, 11, 12, 13, 14, 15);
}

DEVFN unsigned hash_u32(unsigned x) {
  x ^= x >> 17; x *= 0xed5ad4bbu;
  x ^= x >> 11; x *= 0xac4c1b51u;
  x ^= x >> 15; x *= 0x31848babu;
  x ^= x >> 14;
  return x;
}

DEVFN float nrand(unsigned idx) {
  unsigned h1 = hash_u32(idx * 2u + 1u);
  unsigned h2 = hash_u32(idx * 2u + 2u);
  float u1 = (float)(h1 >> 8) * (1.0f / 16777216.0f) + 1e-7f;
  float u2 = (float)(h2 >> 8) * (1.0f / 16777216.0f);
  return sqrtf(-2.0f * __logf(u1)) * __cosf(6.28318530718f * u2);
}

// ---------------------------------------------------------------------------
// Weight pack f32[N][Kin] -> f16[N][Kout] (zero-pad K tail)
// ---------------------------------------------------------------------------
__global__ void k_pack_w(const float* __restrict__ in, half_t* __restrict__ out,
                         int N, int Kin, int Kout) {
  int idx = blockIdx.x * 256 + threadIdx.x;
  if (idx >= N * Kout) return;
  int c = idx % Kout, n = idx / Kout;
  out[idx] = (c < Kin) ? (half_t)in[(size_t)n * Kin + c] : (half_t)0.0f;
}

// ---------------------------------------------------------------------------
// im2col conv1: x[B,T,80] -> A[B*T][416]  (col=i*5+k, row t+k-2, pad K 400->416)
// ---------------------------------------------------------------------------
__global__ void k_im2col_conv1(const float* __restrict__ x, half_t* __restrict__ A,
                               int B, int T) {
  int idx = blockIdx.x * 256 + threadIdx.x;
  int total = B * T * 416;
  if (idx >= total) return;
  int col = idx % 416;
  int bt  = idx / 416;
  int t = bt % T, b = bt / T;
  float v = 0.0f;
  if (col < 400) {
    int i = col / 5, kk = col % 5;
    int tt = t + kk - 2;
    if (tt >= 0 && tt < T) v = x[((size_t)(b * T + tt)) * 80 + i];
  }
  A[idx] = (half_t)v;
}

// ---------------------------------------------------------------------------
// im2col TDNN: act[B,Tin,512] -> A[Mpad][1536] (rows >= B*Tout zero-filled)
// ---------------------------------------------------------------------------
__global__ void k_im2col_tdnn(const half_t* __restrict__ act, half_t* __restrict__ A,
                              int B, int Tin, int Tout, int dil, int Mpad) {
  int idx = blockIdx.x * 256 + threadIdx.x;
  int total = Mpad * 1536;
  if (idx >= total) return;
  int col = idx % 1536;
  int bt  = idx / 1536;
  if (bt >= B * Tout) { A[idx] = (half_t)0.0f; return; }
  int t = bt % Tout, b = bt / Tout;
  int c = col / 3, kk = col % 3;
  A[idx] = act[((size_t)(b * Tin + t + kk * dil)) * 512 + c];
}

// ---------------------------------------------------------------------------
// WMMA GEMM: C[M,N] = A[Mpad,K] * W[N,K]^T  (+bias, optional BN+PReLU)
// Requirements: K % 32 == 0, N % 64 == 0, A has >= gridDim.x*128 valid rows.
// Block 256 threads = 8 waves (4x2), block tile 128x64, wave tile 32x32.
// ---------------------------------------------------------------------------
__global__ __launch_bounds__(256) void k_gemm_wmma(
    const half_t* __restrict__ A, int lda,
    const half_t* __restrict__ W, int ldw,
    int M, int N, int K,
    const float* __restrict__ bias,
    const float* __restrict__ bn_g, const float* __restrict__ bn_b,
    const float* __restrict__ prelu_a,
    half_t* __restrict__ outh, float* __restrict__ outf, int ldo) {
  __shared__ half_t As[128][40];
  __shared__ half_t Ws[64][40];

  int tid  = threadIdx.x;
  int lane = tid & 31;
  int wave = tid >> 5;
  int wm = wave >> 1;
  int wn = wave & 1;
  int bm = blockIdx.x * 128;
  int bn = blockIdx.y * 64;
  int lc = lane & 15;
  int akb = (lane < 16) ? 0 : 8;
  int bkb = (lane < 16) ? 0 : 8;

  int ra = tid >> 1;                 // A stage row 0..127
  int off = (tid & 1) * 16;          // halfs
  const half_t* aptr = A + (size_t)(bm + ra) * lda + off;
  const half_t* wptr = W + (size_t)(bn + (ra & 63)) * ldw + off;

  v8f acc[2][2];
#pragma unroll
  for (int im = 0; im < 2; ++im)
#pragma unroll
    for (int in = 0; in < 2; ++in)
#pragma unroll
      for (int j = 0; j < 8; ++j) acc[im][in][j] = 0.0f;

  for (int k0 = 0; k0 < K; k0 += 32) {
    if (k0 + 32 < K) __builtin_prefetch(aptr + k0 + 32, 0, 1);
#if USE_ASYNC_LDS
    __builtin_amdgcn_global_load_async_to_lds_b128(
        (v4i*)(aptr + k0), (v4i*)&As[ra][off], 0, 0);
    __builtin_amdgcn_global_load_async_to_lds_b128(
        (v4i*)(aptr + k0 + 8), (v4i*)&As[ra][off + 8], 0, 0);
    if (tid < 128) {
      __builtin_amdgcn_global_load_async_to_lds_b128(
          (v4i*)(wptr + k0), (v4i*)&Ws[ra][off], 0, 0);
      __builtin_amdgcn_global_load_async_to_lds_b128(
          (v4i*)(wptr + k0 + 8), (v4i*)&Ws[ra][off + 8], 0, 0);
    }
    __builtin_amdgcn_s_wait_asynccnt(0);
#else
    {
      const uint4* ap = (const uint4*)(aptr + k0);
      uint4 a0 = ap[0], a1 = ap[1];
      *(uint4*)&As[ra][off] = a0;
      *(uint4*)&As[ra][off + 8] = a1;
    }
    if (tid < 128) {
      const uint4* wp = (const uint4*)(wptr + k0);
      uint4 w0 = wp[0], w1 = wp[1];
      *(uint4*)&Ws[ra][off] = w0;
      *(uint4*)&Ws[ra][off + 8] = w1;
    }
#endif
    __syncthreads();

    v16h afr[2], bfr[2];
#pragma unroll
    for (int im = 0; im < 2; ++im) {
      int rr = wm * 32 + im * 16 + lc;
      v8h lo = *(const v8h*)&As[rr][akb];
      v8h hi = *(const v8h*)&As[rr][akb + 16];
      afr[im] = join8(lo, hi);
    }
#pragma unroll
    for (int in = 0; in < 2; ++in) {
      int cc = wn * 32 + in * 16 + lc;
      v8h lo = *(const v8h*)&Ws[cc][bkb];
      v8h hi = *(const v8h*)&Ws[cc][bkb + 16];
      bfr[in] = join8(lo, hi);
    }
#pragma unroll
    for (int im = 0; im < 2; ++im)
#pragma unroll
      for (int in = 0; in < 2; ++in)
        acc[im][in] = wmma_f16(afr[im], bfr[in], acc[im][in]);
    __syncthreads();
  }

  float alpha = prelu_a ? prelu_a[0] : 0.0f;
  float bninv = rsqrtf(1.0f + 1e-5f);
#pragma unroll
  for (int im = 0; im < 2; ++im) {
#pragma unroll
    for (int in = 0; in < 2; ++in) {
      int colg = bn + wn * 32 + in * 16 + lc;
      float bi = bias ? bias[colg] : 0.0f;
      float sc = bn_g ? bn_g[colg] * bninv : 1.0f;
      float sh = bn_b ? bn_b[colg] : 0.0f;
#pragma unroll
      for (int j = 0; j < 8; ++j) {
        int rowg = bm + wm * 32 + im * 16 + j + ((lane < 16) ? 0 : 8);
        if (rowg >= M) continue;
        float vv = acc[im][in][j] + bi;
        vv = vv * sc + sh;
        if (prelu_a) vv = vv > 0.0f ? vv : alpha * vv;
        if (outf) outf[(size_t)rowg * ldo + colg] = vv;
        if (outh) outh[(size_t)rowg * ldo + colg] = (half_t)vv;
      }
    }
  }
}

// ---------------------------------------------------------------------------
// Lorenz: input->chaos projection (512 -> 3)
// ---------------------------------------------------------------------------
__global__ void k_ic_project(const half_t* __restrict__ h1, const float* __restrict__ icw,
                             const float* __restrict__ icb, float* __restrict__ ic, int MT) {
  int idx = blockIdx.x * 256 + threadIdx.x;
  if (idx >= MT * 3) return;
  int j = idx % 3, bt = idx / 3;
  const half_t* row = h1 + (size_t)bt * 512;
  const float* w = icw + j * 512;
  float s = icb[j];
  for (int d = 0; d < 512; ++d) s += (float)row[d] * w[d];
  ic[idx] = s;
}

DEVFN void lorenz_deriv(float x, float y, float z, float cx, float cy, float cz,
                        float& dx, float& dy, float& dz) {
  dx = 10.0f * (y - x) + cx;
  dy = x * (28.0f - z) - y + cy;
  dz = x * y - (8.0f / 3.0f) * z + cz;
}

__global__ __launch_bounds__(32) void k_lorenz_scan(const float* __restrict__ ic,
                                                    const float* __restrict__ coupling,
                                                    float* __restrict__ states, int B, int T) {
  int b = threadIdx.x;
  if (b >= B) return;
  float c[9];
#pragma unroll
  for (int j = 0; j < 9; ++j) c[j] = coupling[j];
  float sx = 0.0f, sy = 0.0f, sz = 0.0f;
  const float hstep = 0.01f;
  for (int t = 0; t < T; ++t) {
    const float* ip = ic + ((size_t)b * T + t) * 3;
    float i0 = ip[0], i1 = ip[1], i2 = ip[2];
    float cx = c[0] * i0 + c[1] * i1 + c[2] * i2;
    float cy = c[3] * i0 + c[4] * i1 + c[5] * i2;
    float cz = c[6] * i0 + c[7] * i1 + c[8] * i2;
    for (int r = 0; r < 10; ++r) {
      float k1x, k1y, k1z, k2x, k2y, k2z, k3x, k3y, k3z, k4x, k4y, k4z;
      lorenz_deriv(sx, sy, sz, cx, cy, cz, k1x, k1y, k1z);
      lorenz_deriv(sx + 0.005f * k1x, sy + 0.005f * k1y, sz + 0.005f * k1z, cx, cy, cz, k2x, k2y, k2z);
      lorenz_deriv(sx + 0.005f * k2x, sy + 0.005f * k2y, sz + 0.005f * k2z, cx, cy, cz, k3x, k3y, k3z);
      lorenz_deriv(sx + 0.01f * k3x, sy + 0.01f * k3y, sz + 0.01f * k3z, cx, cy, cz, k4x, k4y, k4z);
      sx += (hstep / 6.0f) * (k1x + 2.0f * k2x + 2.0f * k3x + k4x);
      sy += (hstep / 6.0f) * (k1y + 2.0f * k2y + 2.0f * k3y + k4y);
      sz += (hstep / 6.0f) * (k1z + 2.0f * k2z + 2.0f * k3z + k4z);
    }
    float* sp = states + ((size_t)b * T + t) * 3;
    sp[0] = sx; sp[1] = sy; sp[2] = sz;
  }
}

__global__ void k_co_expand(const float* __restrict__ states, const float* __restrict__ cow,
                            const float* __restrict__ cob, half_t* __restrict__ h2, int MT) {
  int idx = blockIdx.x * 256 + threadIdx.x;
  if (idx >= MT * 512) return;
  int o = idx % 512, bt = idx / 512;
  const float* s = states + (size_t)bt * 3;
  h2[idx] = (half_t)(s[0] * cow[o * 3 + 0] + s[1] * cow[o * 3 + 1] + s[2] * cow[o * 3 + 2] + cob[o]);
}

// ---------------------------------------------------------------------------
// Bifurcation attention (flash style), 1 wave per (qtile, head, batch).
// ---------------------------------------------------------------------------
__global__ __launch_bounds__(32) void k_attention(
    const half_t* __restrict__ q, const half_t* __restrict__ k,
    const half_t* __restrict__ v, const float* __restrict__ bifp,
    half_t* __restrict__ out, int T) {
  __shared__ half_t Ps[16][40];
  __shared__ half_t Vs[32][136];
  int lane = threadIdx.x;
  int qt = blockIdx.x, hh = blockIdx.y, b = blockIdx.z;
  int hlf = (lane < 16) ? 0 : 1;
  int lc = lane & 15;
  int akb = hlf ? 16 : 0;
  int bkb = hlf ? 16 : 0;

  const half_t* qbase = q + ((size_t)b * T) * 512 + hh * 128;
  const half_t* kbase = k + ((size_t)b * T) * 512 + hh * 128;
  const half_t* vbase = v + ((size_t)b * T) * 512 + hh * 128;

  float pv = bifp[0];
  float r_ = 1.0f / (1.0f + __expf(-pv));
  float bif = r_ * __sinf(3.14159265f * r_);
  float gate = (fabsf(bif - 0.5f) < 0.1f) ? 0.05f : 0.0f;

  v16h zero16;
#pragma unroll
  for (int j = 0; j < 16; ++j) zero16[j] = (half_t)0.0f;

  v16h qa[4];
  int qrow = qt * 16 + lc;
#pragma unroll
  for (int ks = 0; ks < 4; ++ks) {
    const half_t* qp = qbase + (size_t)qrow * 512 + ks * 32;
    v8h lo = *(const v8h*)(qp + akb);
    v8h hi = *(const v8h*)(qp + akb + 8);
    qa[ks] = join8(lo, hi);
  }

  float m[8], l[8];
  v8f O[8];
#pragma unroll
  for (int j = 0; j < 8; ++j) { m[j] = -1e30f; l[j] = 0.0f; }
#pragma unroll
  for (int nt = 0; nt < 8; ++nt)
#pragma unroll
    for (int j = 0; j < 8; ++j) O[nt][j] = 0.0f;

  const float scale = 0.0883883476f;
  unsigned nbase = ((unsigned)(b * 4 + hh)) * 262144u;

  for (int kt = 0; kt < T; kt += 32) {
    {
      int key = kt + lane;
      if (key < T) {
        const uint4* vp = (const uint4*)(vbase + (size_t)key * 512);
#pragma unroll
        for (int c = 0; c < 8; ++c) *(uint4*)&Vs[lane][c * 8] = vp[c];
      } else {
        uint4 z4 = {0u, 0u, 0u, 0u};
#pragma unroll
        for (int c = 0; c < 8; ++c) *(uint4*)&Vs[lane][c * 8] = z4;
      }
    }
    v8f s0, s1;
#pragma unroll
    for (int j = 0; j < 8; ++j) { s0[j] = 0.0f; s1[j] = 0.0f; }
    int key0 = kt + lc, key1 = kt + 16 + lc;
#pragma unroll
    for (int ks = 0; ks < 4; ++ks) {
      v16h b0 = (key0 < T) ? *(const v16h*)(kbase + (size_t)key0 * 512 + ks * 32 + bkb) : zero16;
      v16h b1 = (key1 < T) ? *(const v16h*)(kbase + (size_t)key1 * 512 + ks * 32 + bkb) : zero16;
      s0 = wmma_f16(qa[ks], b0, s0);
      s1 = wmma_f16(qa[ks], b1, s1);
    }
#pragma unroll
    for (int j = 0; j < 8; ++j) {
      int rr = qt * 16 + j + hlf * 8;
      float v0 = (key0 < T) ? s0[j] * scale + gate * nrand(nbase + (unsigned)rr * 512u + (unsigned)key0)
                            : -1e30f;
      float v1 = (key1 < T) ? s1[j] * scale + gate * nrand(nbase + (unsigned)rr * 512u + (unsigned)key1)
                            : -1e30f;
      float mx = fmaxf(v0, v1);
#pragma unroll
      for (int o = 1; o < 16; o <<= 1) mx = fmaxf(mx, __shfl_xor(mx, o, 32));
      float mn = fmaxf(m[j], mx);
      float f = __expf(m[j] - mn);
      float p0 = (key0 < T) ? __expf(v0 - mn) : 0.0f;
      float p1 = (key1 < T) ? __expf(v1 - mn) : 0.0f;
      float ps = p0 + p1;
#pragma unroll
      for (int o = 1; o < 16; o <<= 1) ps += __shfl_xor(ps, o, 32);
      l[j] = l[j] * f + ps;
      m[j] = mn;
#pragma unroll
      for (int nt = 0; nt < 8; ++nt) O[nt][j] *= f;
      Ps[j + hlf * 8][lc]      = (half_t)p0;
      Ps[j + hlf * 8][16 + lc] = (half_t)p1;
    }
    __syncthreads();
    v16h pa;
    {
      v8h lo = *(const v8h*)&Ps[lc][akb];
      v8h hi = *(const v8h*)&Ps[lc][akb + 8];
      pa = join8(lo, hi);
    }
#pragma unroll
    for (int nt = 0; nt < 8; ++nt) {
      v16h bv;
#pragma unroll
      for (int j = 0; j < 16; ++j) bv[j] = Vs[bkb + j][nt * 16 + lc];
      O[nt] = wmma_f16(pa, bv, O[nt]);
    }
    __syncthreads();
  }

  half_t* ob = out + ((size_t)b * T) * 512 + hh * 128;
#pragma unroll
  for (int j = 0; j < 8; ++j) {
    int rr = qt * 16 + j + hlf * 8;
    float inv = 1.0f / l[j];
#pragma unroll
    for (int nt = 0; nt < 8; ++nt)
      ob[(size_t)rr * 512 + nt * 16 + lc] = (half_t)(O[nt][j] * inv);
  }
}

// ---------------------------------------------------------------------------
// attractor positions: pos[m][j] = sum_e proj[m][e] * attr_w[e][j]
// ---------------------------------------------------------------------------
__global__ void k_pos(const float* __restrict__ proj, const float* __restrict__ attrw,
                      float* __restrict__ pos, int MS) {
  int idx = blockIdx.x * 256 + threadIdx.x;
  if (idx >= MS * 3) return;
  int j = idx % 3, mrow = idx / 3;
  const float* p = proj + (size_t)mrow * 192;
  float s = 0.0f;
  for (int e = 0; e < 192; ++e) s += p[e] * attrw[e * 3 + j];
  pos[idx] = s;
}

// ---------------------------------------------------------------------------
// Strange-attractor pooling, one block per batch element.
// ---------------------------------------------------------------------------
__global__ __launch_bounds__(256) void k_pool(const half_t* __restrict__ h,
                                              const float* __restrict__ pos,
                                              float* __restrict__ pooled, int S) {
  __shared__ float wsh[512];
  __shared__ float red[256];
  __shared__ float center[3];
  int b = blockIdx.x, tid = threadIdx.x;

  for (int j = 0; j < 3; ++j) {
    float s = 0.0f;
    for (int i = tid; i < S; i += 256) s += pos[((size_t)b * S + i) * 3 + j];
    red[tid] = s;
    __syncthreads();
    for (int o = 128; o > 0; o >>= 1) { if (tid < o) red[tid] += red[tid + o]; __syncthreads(); }
    if (tid == 0) center[j] = red[0] / (float)S;
    __syncthreads();
  }
  float lmax = -1e30f;
  for (int i = tid; i < S; i += 256) {
    const float* pp = pos + ((size_t)b * S + i) * 3;
    float dx = pp[0] - center[0], dy = pp[1] - center[1], dz = pp[2] - center[2];
    float d = sqrtf(dx * dx + dy * dy + dz * dz);
    wsh[i] = -d;
    lmax = fmaxf(lmax, -d);
  }
  red[tid] = lmax;
  __syncthreads();
  for (int o = 128; o > 0; o >>= 1) { if (tid < o) red[tid] = fmaxf(red[tid], red[tid + o]); __syncthreads(); }
  float mx = red[0];
  __syncthreads();
  float lsum = 0.0f;
  for (int i = tid; i < S; i += 256) {
    float e = __expf(wsh[i] - mx);
    wsh[i] = e;
    lsum += e;
  }
  red[tid] = lsum;
  __syncthreads();
  for (int o = 128; o > 0; o >>= 1) { if (tid < o) red[tid] += red[tid + o]; __syncthreads(); }
  float tot = red[0];
  __syncthreads();
  for (int i = tid; i < S; i += 256) wsh[i] /= tot;
  __syncthreads();
  for (int d = tid; d < 512; d += 256) {
    float s = 0.0f;
    for (int i = 0; i < S; ++i) s += (float)h[((size_t)b * S + i) * 512 + d] * wsh[i];
    pooled[b * 512 + d] = s;
  }
}

// ---------------------------------------------------------------------------
// Small dense tail
// ---------------------------------------------------------------------------
__global__ void k_dense(const float* __restrict__ in, const float* __restrict__ W,
                        const float* __restrict__ bias,
                        const float* __restrict__ bn_g, const float* __restrict__ bn_b,
                        const float* __restrict__ prelu_a,
                        float* __restrict__ out, int M, int N, int K) {
  int idx = blockIdx.x * 256 + threadIdx.x;
  if (idx >= M * N) return;
  int n = idx % N, mrow = idx / N;
  const float* a = in + (size_t)mrow * K;
  const float* w = W + (size_t)n * K;
  float s = bias ? bias[n] : 0.0f;
  for (int kk = 0; kk < K; ++kk) s += a[kk] * w[kk];
  if (bn_g) s = s * (bn_g[n] * rsqrtf(1.0f + 1e-5f)) + bn_b[n];
  if (prelu_a) { float al = prelu_a[0]; s = s > 0.0f ? s : al * s; }
  out[idx] = s;
}

// ---------------------------------------------------------------------------
// Host orchestration
// ---------------------------------------------------------------------------
static inline int ceil_div(int a, int b) { return (a + b - 1) / b; }

extern "C" void kernel_launch(void* const* d_in, const int* in_sizes, int n_in,
                              void* d_out, int out_size, void* d_ws, size_t ws_size,
                              hipStream_t stream) {
  (void)in_sizes; (void)n_in; (void)out_size; (void)ws_size;
  const int B = 32, T = 400;
  const int MT = B * T;                  // 12800 (multiple of 128)
  const int T1 = 398, T2 = 394, T3 = 388;
  const int M1 = B * T1, M2 = B * T2, M3 = B * T3;
  const int MPAD = 12800;

  const float* x        = (const float*)d_in[0];
  const float* conv1_w  = (const float*)d_in[1];
  const float* conv1_b  = (const float*)d_in[2];
  const float* bn1_g    = (const float*)d_in[3];
  const float* bn1_b    = (const float*)d_in[4];
  const float* prelu1   = (const float*)d_in[5];
  const float* ic_w     = (const float*)d_in[6];
  const float* ic_b     = (const float*)d_in[7];
  const float* co_w     = (const float*)d_in[8];
  const float* co_b     = (const float*)d_in[9];
  const float* coupling = (const float*)d_in[10];
  const float* q_w      = (const float*)d_in[11];
  const float* q_b      = (const float*)d_in[12];
  const float* k_w      = (const float*)d_in[13];
  const float* k_b      = (const float*)d_in[14];
  const float* v_w      = (const float*)d_in[15];
  const float* v_b      = (const float*)d_in[16];
  const float* bif_p    = (const float*)d_in[17];
  const float* tdnn_w[3] = {(const float*)d_in[18], (const float*)d_in[23], (const float*)d_in[28]};
  const float* tdnn_b[3] = {(const float*)d_in[19], (const float*)d_in[24], (const float*)d_in[29]};
  const float* bn_t_g[3] = {(const float*)d_in[20], (const float*)d_in[25], (const float*)d_in[30]};
  const float* bn_t_b[3] = {(const float*)d_in[21], (const float*)d_in[26], (const float*)d_in[31]};
  const float* prelu_t[3]= {(const float*)d_in[22], (const float*)d_in[27], (const float*)d_in[32]};
  const float* proj_w   = (const float*)d_in[33];
  const float* proj_b   = (const float*)d_in[34];
  const float* attr_w   = (const float*)d_in[35];
  const float* fc1_w    = (const float*)d_in[36];
  const float* fc1_b    = (const float*)d_in[37];
  const float* bn_fc1_g = (const float*)d_in[38];
  const float* bn_fc1_b = (const float*)d_in[39];
  const float* prelu_fc1= (const float*)d_in[40];
  const float* cls_w    = (const float*)d_in[41];
  const float* cls_b    = (const float*)d_in[42];

  char* p = (char*)d_ws;
  auto alloc = [&](size_t bytes) -> void* {
    void* r = (void*)p;
    p += (bytes + 255) & ~(size_t)255;
    return r;
  };
  half_t* wconv1 = (half_t*)alloc((size_t)512 * 416 * 2);
  half_t* wq     = (half_t*)alloc((size_t)512 * 512 * 2);
  half_t* wk     = (half_t*)alloc((size_t)512 * 512 * 2);
  half_t* wv     = (half_t*)alloc((size_t)512 * 512 * 2);
  half_t* wt[3];
  for (int i = 0; i < 3; ++i) wt[i] = (half_t*)alloc((size_t)512 * 1536 * 2);
  half_t* wproj  = (half_t*)alloc((size_t)192 * 512 * 2);
  half_t* h1     = (half_t*)alloc((size_t)MT * 512 * 2);
  half_t* h2     = (half_t*)alloc((size_t)MT * 512 * 2);
  half_t* qh     = (half_t*)alloc((size_t)MT * 512 * 2);
  half_t* kh     = (half_t*)alloc((size_t)MT * 512 * 2);
  half_t* vh     = (half_t*)alloc((size_t)MT * 512 * 2);
  half_t* h3     = (half_t*)alloc((size_t)MT * 512 * 2);
  half_t* t1     = (half_t*)alloc((size_t)M1 * 512 * 2);
  half_t* t2     = (half_t*)alloc((size_t)M2 * 512 * 2);
  half_t* t3     = (half_t*)alloc((size_t)M3 * 512 * 2);
  half_t* Acol   = (half_t*)alloc((size_t)MPAD * 1536 * 2);
  float*  icbuf  = (float*)alloc((size_t)MT * 3 * 4);
  float*  states = (float*)alloc((size_t)MT * 3 * 4);
  float*  projf  = (float*)alloc((size_t)M3 * 192 * 4);
  float*  posf   = (float*)alloc((size_t)M3 * 3 * 4);
  float*  pooled = (float*)alloc((size_t)B * 512 * 4);
  float*  emb0   = (float*)alloc((size_t)B * 192 * 4);

  float* out_emb    = (float*)d_out;
  float* out_logits = out_emb + B * 192;

  auto pack = [&](const float* src, half_t* dst, int N, int Kin, int Kout) {
    k_pack_w<<<ceil_div(N * Kout, 256), 256, 0, stream>>>(src, dst, N, Kin, Kout);
  };
  pack(conv1_w, wconv1, 512, 400, 416);
  pack(q_w, wq, 512, 512, 512);
  pack(k_w, wk, 512, 512, 512);
  pack(v_w, wv, 512, 512, 512);
  for (int i = 0; i < 3; ++i) pack(tdnn_w[i], wt[i], 512, 1536, 1536);
  pack(proj_w, wproj, 192, 512, 512);

  // conv1 as GEMM (+BN+PReLU), K padded to 416
  k_im2col_conv1<<<ceil_div(MT * 416, 256), 256, 0, stream>>>(x, Acol, B, T);
  k_gemm_wmma<<<dim3(MT / 128, 512 / 64), 256, 0, stream>>>(
      Acol, 416, wconv1, 416, MT, 512, 416,
      conv1_b, bn1_g, bn1_b, prelu1, h1, nullptr, 512);

  // Lorenz chaotic module
  k_ic_project<<<ceil_div(MT * 3, 256), 256, 0, stream>>>(h1, ic_w, ic_b, icbuf, MT);
  k_lorenz_scan<<<1, 32, 0, stream>>>(icbuf, coupling, states, B, T);
  k_co_expand<<<ceil_div(MT * 512, 256), 256, 0, stream>>>(states, co_w, co_b, h2, MT);

  // QKV projections
  dim3 gq(MT / 128, 512 / 64);
  k_gemm_wmma<<<gq, 256, 0, stream>>>(h2, 512, wq, 512, MT, 512, 512,
                                      q_b, nullptr, nullptr, nullptr, qh, nullptr, 512);
  k_gemm_wmma<<<gq, 256, 0, stream>>>(h2, 512, wk, 512, MT, 512, 512,
                                      k_b, nullptr, nullptr, nullptr, kh, nullptr, 512);
  k_gemm_wmma<<<gq, 256, 0, stream>>>(h2, 512, wv, 512, MT, 512, 512,
                                      v_b, nullptr, nullptr, nullptr, vh, nullptr, 512);

  // bifurcation attention
  k_attention<<<dim3(T / 16, 4, B), 32, 0, stream>>>(qh, kh, vh, bif_p, h3, T);

  // TDNN stack
  const half_t* tin[3] = {h3, t1, t2};
  half_t* tout[3] = {t1, t2, t3};
  const int tins[3]  = {T, T1, T2};
  const int touts[3] = {T1, T2, T3};
  const int dils[3]  = {1, 2, 3};
  for (int i = 0; i < 3; ++i) {
    int Mi = B * touts[i];
    int Mg = ceil_div(Mi, 128) * 128;
    k_im2col_tdnn<<<ceil_div(Mg * 1536, 256), 256, 0, stream>>>(
        tin[i], Acol, B, tins[i], touts[i], dils[i], Mg);
    k_gemm_wmma<<<dim3(Mg / 128, 512 / 64), 256, 0, stream>>>(
        Acol, 1536, wt[i], 1536, Mi, 512, 1536,
        tdnn_b[i], bn_t_g[i], bn_t_b[i], prelu_t[i], tout[i], nullptr, 512);
  }

  // strange-attractor pooling
  k_gemm_wmma<<<dim3(M3 / 128, 192 / 64), 256, 0, stream>>>(
      t3, 512, wproj, 512, M3, 192, 512,
      proj_b, nullptr, nullptr, nullptr, nullptr, projf, 192);
  k_pos<<<ceil_div(M3 * 3, 256), 256, 0, stream>>>(projf, attr_w, posf, M3);
  k_pool<<<B, 256, 0, stream>>>(t3, posf, pooled, T3);

  // tail
  k_dense<<<ceil_div(B * 192, 256), 256, 0, stream>>>(
      pooled, proj_w, proj_b, nullptr, nullptr, nullptr, emb0, B, 192, 512);
  k_dense<<<ceil_div(B * 192, 256), 256, 0, stream>>>(
      emb0, fc1_w, fc1_b, bn_fc1_g, bn_fc1_b, prelu_fc1, out_emb, B, 192, 192);
  k_dense<<<ceil_div(B * 1211, 256), 256, 0, stream>>>(
      out_emb, cls_w, cls_b, nullptr, nullptr, nullptr, out_logits, B, 1211, 192);
}
